// MS_SSIM_3384434230162
// MI455X (gfx1250) — matrix operand order repaired and checked by reading, not compiled, not run on hardware
//
#include <hip/hip_runtime.h>
#include <math.h>

typedef _Float16 v16h  __attribute__((ext_vector_type(16)));
typedef float    v8f   __attribute__((ext_vector_type(8)));
typedef float    f4raw __attribute__((ext_vector_type(4)));   // trivial, 16B aligned

#define SSIM_C1 6.5025f     // (0.01*255)^2
#define SSIM_C2 58.5225f    // (0.03*255)^2

#define PLW 40   // f16 plane row stride (halfs): 16B-aligned rows, bank-spread
#define SHW 40   // f16 horizontal-result column stride (halfs)

// ---------------------------------------------------------------------------
__global__ void msssim_init_kernel(double* acc) {
    if (threadIdx.x < 16) acc[threadIdx.x] = 0.0;
}

// ---------------------------------------------------------------------------
// One wave per 16x16 output tile of one (batch, channel) plane.
// Fused: halo load -> five f16 product planes in LDS -> 5 separable 11x11
// Gaussian convs as banded-weight WMMAs -> pointwise SSIM/MCS -> wave-reduced
// double atomics -> exact fp32 2x2 avg-pool for the next level.
// ---------------------------------------------------------------------------
__global__ __launch_bounds__(32) void msssim_level_kernel(
    const float* __restrict__ in1, const float* __restrict__ in2,
    float* __restrict__ out1, float* __restrict__ out2,
    double* __restrict__ acc, int H, int W, int level, int tilesX)
{
    // f16 product planes: x, y, xx, yy, xy.  32 rows x PLW halfs, zero padded.
    __shared__ __align__(16) _Float16 pl[5][32][PLW];
    // horizontal conv results per plane, transposed: [col][k], k = 0..31
    __shared__ __align__(16) _Float16 sh[5][16][SHW];
    // raw f32 tile for exact 2x2 pooling
    __shared__ float tp1[16][17], tp2[16][17];

    const int lane = threadIdx.x;          // 0..31 (wave32)
    const int hl   = lane >> 4;            // lane half
    const int r16  = lane & 15;

    const int x0 = (blockIdx.x % tilesX) << 4;
    const int y0 = (blockIdx.x / tilesX) << 4;
    const int plane = blockIdx.z * 50 + blockIdx.y;
    const size_t pbase = (size_t)plane * H * W;

    // Zero-fill the product planes (pad regions must be finite/zero).
    {
        f4raw z = {0.f, 0.f, 0.f, 0.f};
        f4raw* pz = (f4raw*)&pl[0][0][0];
        const int n16 = (int)(sizeof(pl) / 16);
        for (int i = lane; i < n16; i += 32) pz[i] = z;
    }

    // Gaussian window: ws=11, sigma=1.5 at every level (min dim >= 16)
    float g[11];
    {
        float s = 0.f;
        #pragma unroll
        for (int i = 0; i < 11; ++i) {
            float d = (float)(i - 5);
            g[i] = expf(-d * d / 4.5f);
            s += g[i];
        }
        #pragma unroll
        for (int i = 0; i < 11; ++i) g[i] /= s;
    }

    // Halo load (26x26, zero padded like lax.conv) + build f16 product planes
    // and the raw f32 pooling tile.
    for (int idx = lane; idx < 26 * 13; idx += 32) {
        int r  = idx / 13;
        int c0 = 2 * (idx - r * 13);
        int gy = y0 - 5 + r;
        #pragma unroll
        for (int t = 0; t < 2; ++t) {
            int c  = c0 + t;
            int gx = x0 - 5 + c;
            float a = 0.f, b = 0.f;
            if (gy >= 0 && gy < H && gx >= 0 && gx < W) {
                size_t o = pbase + (size_t)gy * W + gx;
                a = in1[o];
                b = in2[o];
            }
            pl[0][r][c] = (_Float16)a;
            pl[1][r][c] = (_Float16)b;
            pl[2][r][c] = (_Float16)(a * a);
            pl[3][r][c] = (_Float16)(b * b);
            pl[4][r][c] = (_Float16)(a * b);
            if (r >= 5 && r < 21 && c >= 5 && c < 21) {
                tp1[r - 5][c - 5] = a;
                tp2[r - 5][c - 5] = b;
            }
        }
    }
    __syncthreads();

    // Banded Gaussian weight fragments (all lanes, branch-free construction).
    // Horizontal B (32x16): B[k][n] = g[k-n].  Vertical A (16x32): A[m][k]=g[k-m].
    v16h Bg, Ag;
    #pragma unroll
    for (int i = 0; i < 8; ++i) {
        int kb = ((i < 4) ? 2 * i : 8 + 2 * i) + 8 * hl;
        int d0 = kb - r16, d1 = kb + 1 - r16;
        Bg[2*i]   = (d0 >= 0 && d0 <= 10) ? (_Float16)g[d0] : (_Float16)0.f;
        Bg[2*i+1] = (d1 >= 0 && d1 <= 10) ? (_Float16)g[d1] : (_Float16)0.f;
        int ka = 2 * i + 16 * hl;
        int e0 = ka - r16, e1 = ka + 1 - r16;
        Ag[2*i]   = (e0 >= 0 && e0 <= 10) ? (_Float16)g[e0] : (_Float16)0.f;
        Ag[2*i+1] = (e1 >= 0 && e1 <= 10) ? (_Float16)g[e1] : (_Float16)0.f;
    }

    const v8f zero = {};

    // ---- horizontal pass: row-blocks 0..15 and 16..31 (16..31: rows 26+ are
    // padded zeros; the vertical band never weights them anyway).
    #pragma unroll
    for (int blk = 0; blk < 2; ++blk) {
        const int rowOff = blk * 16;
        const int r = rowOff + r16;
        #pragma unroll
        for (int p = 0; p < 5; ++p) {
            union { v16h h; f4raw f[2]; } A;
            const _Float16* base = &pl[p][r][8 * hl];
            A.f[0] = *(const f4raw*)(const void*)(base);
            A.f[1] = *(const f4raw*)(const void*)(base + 16);
            v8f hD = __builtin_amdgcn_wmma_f32_16x16x32_f16(
                false, A.h, false, Bg, (short)0, zero, false, false);
            union { f4raw f; _Float16 hx[8]; } st;
            #pragma unroll
            for (int i = 0; i < 8; ++i) st.hx[i] = (_Float16)hD[i];
            *(f4raw*)(void*)&sh[p][r16][rowOff + 8 * hl] = st.f;
        }
    }
    __syncthreads();

    // ---- vertical pass: banded weights x 32 H-rows (rows >= 26 are zero)
    v8f res[5];
    #pragma unroll
    for (int p = 0; p < 5; ++p) {
        union { v16h h; f4raw f[2]; } Bv;
        const _Float16* base = &sh[p][r16][16 * hl];
        Bv.f[0] = *(const f4raw*)(const void*)(base);
        Bv.f[1] = *(const f4raw*)(const void*)(base + 8);
        res[p] = __builtin_amdgcn_wmma_f32_16x16x32_f16(
            false, Ag, false, Bv.h, (short)0, zero, false, false);
    }

    // ---- pointwise SSIM / MCS over this lane's 8 elements
    float ssum = 0.f, msum = 0.f;
    #pragma unroll
    for (int i = 0; i < 8; ++i) {
        float m1 = res[0][i], m2 = res[1][i];
        float e11 = res[2][i], e22 = res[3][i], e12 = res[4][i];
        float m1s = m1 * m1, m2s = m2 * m2, m12 = m1 * m2;
        float V1 = 2.f * (e12 - m12) + SSIM_C2;
        float V2 = (e11 - m1s) + (e22 - m2s) + SSIM_C2;
        ssum += (2.f * m12 + SSIM_C1) * V1 / ((m1s + m2s + SSIM_C1) * V2);
        msum += V1 / V2;
    }
    #pragma unroll
    for (int off = 16; off > 0; off >>= 1) {
        ssum += __shfl_xor(ssum, off, 32);
        msum += __shfl_xor(msum, off, 32);
    }
    if (lane == 0) {
        atomicAdd(&acc[2 * level],     (double)ssum);
        atomicAdd(&acc[2 * level + 1], (double)msum);
    }

    // ---- exact fp32 2x2 avg-pool for the next level
    if (out1 != nullptr) {
        int Hn = H >> 1, Wn = W >> 1;
        size_t nbase = (size_t)plane * Hn * Wn;
        #pragma unroll
        for (int t = 0; t < 2; ++t) {
            int idx = lane + t * 32;
            int ty = idx >> 3, tx = idx & 7;
            int r = 2 * ty, c = 2 * tx;
            float p1 = 0.25f * (tp1[r][c] + tp1[r][c+1] + tp1[r+1][c] + tp1[r+1][c+1]);
            float p2 = 0.25f * (tp2[r][c] + tp2[r][c+1] + tp2[r+1][c] + tp2[r+1][c+1]);
            size_t o = nbase + (size_t)((y0 >> 1) + ty) * Wn + (x0 >> 1) + tx;
            out1[o] = p1;
            out2[o] = p2;
        }
    }
}

// ---------------------------------------------------------------------------
__global__ void msssim_finalize_kernel(const double* __restrict__ acc,
                                       float* __restrict__ out) {
    if (threadIdx.x == 0 && blockIdx.x == 0) {
        const double w[5] = {0.0448, 0.2856, 0.3001, 0.2363, 0.1333};
        double v = 1.0;
        for (int l = 0; l < 5; ++l) {
            int dim = 256 >> l;
            double n = 4.0 * 50.0 * (double)dim * (double)dim;
            double ssim = acc[2 * l] / n;
            double mcs  = acc[2 * l + 1] / n;
            v *= (l < 4) ? pow(mcs, w[l]) : pow(ssim, w[4]);
        }
        out[0] = (float)v;
    }
}

// ---------------------------------------------------------------------------
extern "C" void kernel_launch(void* const* d_in, const int* in_sizes, int n_in,
                              void* d_out, int out_size, void* d_ws, size_t ws_size,
                              hipStream_t stream) {
    (void)in_sizes; (void)n_in; (void)out_size; (void)ws_size;

    static const int HS[5] = {256, 128, 64, 32, 16};

    char*   wsb = (char*)d_ws;
    double* acc = (double*)wsb;                 // 16 doubles (128 B)
    float*  lvl = (float*)(wsb + 128);          // pooled image buffers

    const float* A[5];
    const float* B[5];
    float* bufA[5] = {nullptr, nullptr, nullptr, nullptr, nullptr};
    float* bufB[5] = {nullptr, nullptr, nullptr, nullptr, nullptr};

    A[0] = (const float*)d_in[0];
    B[0] = (const float*)d_in[1];
    size_t off = 0;
    for (int l = 1; l < 5; ++l) {
        size_t n = 200ull * HS[l] * HS[l];      // 4 batch * 50 chan * H * W
        bufA[l] = lvl + off; off += n;
        bufB[l] = lvl + off; off += n;
        A[l] = bufA[l];
        B[l] = bufB[l];
    }

    msssim_init_kernel<<<1, 32, 0, stream>>>(acc);

    for (int l = 0; l < 5; ++l) {
        int Hd = HS[l];
        int tilesX = Hd / 16;
        dim3 grid(tilesX * tilesX, 50, 4);
        float* o1 = (l < 4) ? bufA[l + 1] : nullptr;
        float* o2 = (l < 4) ? bufB[l + 1] : nullptr;
        msssim_level_kernel<<<grid, 32, 0, stream>>>(
            A[l], B[l], o1, o2, acc, Hd, Hd, l, tilesX);
    }

    msssim_finalize_kernel<<<1, 32, 0, stream>>>(acc, (float*)d_out);
}
